// WaveletBlock_25580825215537
// MI455X (gfx1250) — matrix operand order repaired and checked by reading, compiled, not run on hardware
//
#include <hip/hip_runtime.h>
#include <hip/hip_bf16.h>

typedef __attribute__((ext_vector_type(16))) _Float16 v16h;
typedef __attribute__((ext_vector_type(8)))  _Float16 v8h;
typedef __attribute__((ext_vector_type(4)))  _Float16 v4h;
typedef __attribute__((ext_vector_type(8)))  float    v8f;

#define WMMA_F16(a, b, c) \
  __builtin_amdgcn_wmma_f32_16x16x32_f16(false, (a), false, (b), (short)0, (c), false, false)

// ---------------------------------------------------------------------------
// Fragment layouts per CDNA5 ISA 7.12.2 (wave32, 16x16x32 f16):
// A 16x32: lane m=lane&15, hi=lane>>4; halves 0-7 -> K=hi*8+i, 8-15 -> K=hi*8+16+i
// B 32x16: lane n=lane&15, hi=lane>>4; half h -> K=hi*8+(h&7)+16*(h>>3)
// C/D f32: vgpr r, lane: row m = r+8*(lane>>4), col n = lane&15
// B operands PRE-PACKED: block(ntile,kstep) = 32 lanes x 16 halves contiguous
// ---------------------------------------------------------------------------
__device__ __forceinline__ v16h fragA_f16(const _Float16* base, int lda, int kk) {
  const int lane = threadIdx.x & 31;
  const int m = lane & 15, hi = lane >> 4;
  const _Float16* p = base + (size_t)m * lda + kk + hi * 8;
  v8h lo = *(const v8h*)p;
  v8h hv = *(const v8h*)(p + 16);
  v16h a;
#pragma unroll
  for (int i = 0; i < 8; ++i) { a[i] = lo[i]; a[8 + i] = hv[i]; }
  return a;
}

__device__ __forceinline__ v16h fragA_f16_ptr(const _Float16* prow, bool valid) {
  const int hi = (threadIdx.x & 31) >> 4;
  v16h a;
  if (valid) {
    const _Float16* p = prow + hi * 8;
    v8h lo = *(const v8h*)p;
    v8h hv = *(const v8h*)(p + 16);
#pragma unroll
    for (int i = 0; i < 8; ++i) { a[i] = lo[i]; a[8 + i] = hv[i]; }
  } else {
#pragma unroll
    for (int i = 0; i < 16; ++i) a[i] = (_Float16)0.f;
  }
  return a;
}

__device__ __forceinline__ v16h fragA_f32_ptr(const float* __restrict__ prow) {
  const int hi = (threadIdx.x & 31) >> 4;
  const float* p = prow + hi * 8;
  v16h a;
#pragma unroll
  for (int i = 0; i < 8; ++i) a[i] = (_Float16)p[i];
#pragma unroll
  for (int i = 0; i < 8; ++i) a[8 + i] = (_Float16)p[16 + i];
  return a;
}

__device__ __forceinline__ v16h fragB_pk(const _Float16* __restrict__ blk) {
  return *(const v16h*)(blk + (threadIdx.x & 31) * 16);
}

// ---------------------------------------------------------------------------
// gfx1250 async global->LDS DMA (ASYNCcnt), see CDNA5 ISA
// ---------------------------------------------------------------------------
__device__ __forceinline__ unsigned lds_off32(const void* p) {
  return (unsigned)(size_t)p;  // generic addr low 32 bits == LDS byte offset
}
__device__ __forceinline__ void async_g2l_b128(unsigned lds_addr, const void* gptr) {
  asm volatile("global_load_async_to_lds_b128 %0, %1, off"
               :: "v"(lds_addr), "v"((unsigned long long)gptr) : "memory");
}
__device__ __forceinline__ void wait_async_all() {
  asm volatile("s_wait_asynccnt 0x0" ::: "memory");
}
// stage nch 16-byte chunks from contiguous global region into LDS (block-wide)
__device__ __forceinline__ void stage_async(const _Float16* g, unsigned lds_base, int nch) {
  for (int c = threadIdx.x; c < nch; c += 256)
    async_g2l_b128(lds_base + c * 16, (const char*)g + (size_t)c * 16);
}

// ---------------------------------------------------------------------------
// Pack fp32 weight B(k,n)=src[n*snn + k*skn + tap] into fragment-layout f16.
// ---------------------------------------------------------------------------
__global__ void k_packB(const float* __restrict__ src, _Float16* __restrict__ dst,
                        int KS, int NT, long skn, long snn, int ntaps) {
  long idx = (long)blockIdx.x * 256 + threadIdx.x;
  long per = (long)NT * KS * 512;
  if (idx >= per * ntaps) return;
  int tap = (int)(idx / per);
  long r = idx % per;
  int h = (int)(r & 15);
  int lane = (int)((r >> 4) & 31);
  long r2 = r >> 9;
  int ks = (int)(r2 % KS), nt = (int)(r2 / KS);
  int n = nt * 16 + (lane & 15);
  int k = ks * 32 + (lane >> 4) * 8 + (h & 7) + (h >> 3) * 16;
  dst[idx] = (_Float16)src[(long)n * snn + (long)k * skn + tap];
}

__global__ void k_cvt4(const float4* __restrict__ in, v4h* __restrict__ out, long n4) {
  long i = (long)blockIdx.x * 256 + threadIdx.x;
  if (i >= n4) return;
  float4 v = in[i];
  v4h o; o[0] = (_Float16)v.x; o[1] = (_Float16)v.y; o[2] = (_Float16)v.z; o[3] = (_Float16)v.w;
  out[i] = o;
}

// ---------------------------------------------------------------------------
// 1) q = (x @ q_w.T + q_b)*0.125 -> f16 (b*8+head, L, 64). M=65536 N=512 K=512
//    A tile staged to LDS via async DMA, each wave serves 4 N-tiles.
// ---------------------------------------------------------------------------
__global__ void k_qproj(const _Float16* __restrict__ x16, const _Float16* __restrict__ wpk,
                        const float* __restrict__ qb, _Float16* __restrict__ qout) {
  __shared__ __align__(16) _Float16 atile[16 * 512];
  const int wave = threadIdx.x >> 5, lane = threadIdx.x & 31;
  const int mt = blockIdx.x;
  stage_async(x16 + (size_t)mt * 8192, lds_off32(atile), 1024);
  wait_async_all();
  __syncthreads();
  const int hi = lane >> 4;
#pragma unroll
  for (int g = 0; g < 4; ++g) {
    const int nt = g * 8 + wave;
    const _Float16* Bp = wpk + (size_t)nt * 8192;
    v8f c = {};
#pragma unroll 4
    for (int ks = 0; ks < 16; ++ks) {
      v16h a = fragA_f16(atile, 512, ks * 32);
      v16h b = fragB_pk(Bp + ks * 512);
      c = WMMA_F16(a, b, c);
    }
    const int n = nt * 16 + (lane & 15);
    const float bias = qb[n];
    const int head = n >> 6, d = n & 63;
#pragma unroll
    for (int r = 0; r < 8; ++r) {
      int m = mt * 16 + r + 8 * hi;
      int b_ = m >> 12, l = m & 4095;
      qout[((size_t)(b_ * 8 + head) * 4096 + l) * 64 + d] = (_Float16)((c[r] + bias) * 0.125f);
    }
  }
}

// ---------------------------------------------------------------------------
// 2) 1x1 reduce conv + bn1 + relu -> xred fp32 NHWC (B,64,64,128)
// ---------------------------------------------------------------------------
__global__ void k_redconv(const _Float16* __restrict__ x16, const _Float16* __restrict__ wpk,
                          const float* __restrict__ rb, const float* __restrict__ g,
                          const float* __restrict__ bb, const float* __restrict__ mm,
                          const float* __restrict__ vv, float* __restrict__ xred) {
  __shared__ __align__(16) _Float16 atile[16 * 512];
  const int wave = threadIdx.x >> 5, lane = threadIdx.x & 31;
  const int mt = blockIdx.x, nt = wave;  // 8 N-tiles
  stage_async(x16 + (size_t)mt * 8192, lds_off32(atile), 1024);
  wait_async_all();
  __syncthreads();
  const _Float16* Bp = wpk + (size_t)nt * 8192;
  v8f c = {};
#pragma unroll 4
  for (int ks = 0; ks < 16; ++ks) {
    v16h a = fragA_f16(atile, 512, ks * 32);
    v16h b = fragB_pk(Bp + ks * 512);
    c = WMMA_F16(a, b, c);
  }
  const int n = nt * 16 + (lane & 15), hi = lane >> 4;
  const float inv = g[n] * rsqrtf(vv[n] + 1e-5f);
  const float sh = rb[n] * inv + bb[n] - mm[n] * inv;
#pragma unroll
  for (int r = 0; r < 8; ++r) {
    float val = c[r] * inv + sh;
    xred[(size_t)(mt * 16 + r + 8 * hi) * 128 + n] = val > 0.f ? val : 0.f;
  }
}

// ---------------------------------------------------------------------------
// 3) Haar DWT: xred fp32 -> dwt0 f16 NHWC (B,32,32,512) [ll,lh,hl,hh]
// ---------------------------------------------------------------------------
__global__ void k_dwt(const float* __restrict__ xred, _Float16* __restrict__ out) {
  int idx = blockIdx.x * 256 + threadIdx.x;
  int cr = idx & 127; int p = idx >> 7;
  int j = p & 31; p >>= 5; int i = p & 31; int b = p >> 5;
  const float* s = xred + ((size_t)((b * 64 + 2 * i) * 64 + 2 * j)) * 128 + cr;
  float x00 = s[0], x01 = s[128], x10 = s[64 * 128], x11 = s[64 * 128 + 128];
  _Float16* o = out + ((size_t)((b * 32 + i) * 32 + j)) * 512 + cr;
  o[0]   = (_Float16)(0.5f * (x00 + x01 + x10 + x11));
  o[128] = (_Float16)(0.5f * (x00 + x01 - x10 - x11));
  o[256] = (_Float16)(0.5f * (x00 - x01 + x10 - x11));
  o[384] = (_Float16)(0.5f * (x00 - x01 - x10 + x11));
}

// ---------------------------------------------------------------------------
// 4) 3x3 conv + bn2 + relu (implicit GEMM), M=16384 N=512 K=9*512, pad=1.
//    Halo patch (3 rows x 18 cols x 512ch f16 = 54KB) staged via async DMA;
//    each wave serves 4 N-tiles from it.
// ---------------------------------------------------------------------------
__global__ void k_conv3(const _Float16* __restrict__ xin, const _Float16* __restrict__ wpk,
                        const float* __restrict__ fb, const float* __restrict__ g,
                        const float* __restrict__ bb, const float* __restrict__ mm,
                        const float* __restrict__ vv, float* __restrict__ yout) {
  __shared__ __align__(16) _Float16 patch[3 * 18 * 512];
  const int wave = threadIdx.x >> 5, lane = threadIdx.x & 31;
  const int mt = blockIdx.x;
  const int mrow = lane & 15, hi = lane >> 4;
  const int t0 = mt * 16;
  const int b_ = t0 >> 10, oh = (t0 >> 5) & 31, ow0 = t0 & 31;
  // stage: chunk c -> row r3 (0..2), col (0..17), 16B sub-chunk cic (0..63)
  const unsigned pbase = lds_off32(patch);
  for (int c = threadIdx.x; c < 3456; c += 256) {
    int r3 = c / 1152, rem = c % 1152;
    int col = rem >> 6, cic = rem & 63;
    int iy = oh - 1 + r3, ix = ow0 - 1 + col;
    if (iy >= 0 && iy < 32 && ix >= 0 && ix < 32)
      async_g2l_b128(pbase + c * 16,
                     xin + ((size_t)((b_ * 32 + iy) * 32 + ix)) * 512 + cic * 8);
  }
  wait_async_all();
  __syncthreads();

  const int ow = ow0 + mrow;
  for (int gq = 0; gq < 4; ++gq) {
    const int nt = gq * 8 + wave;
    v8f c = {};
    for (int dy = -1; dy <= 1; ++dy) {
      const int iy = oh + dy;
      const bool rowok = (iy >= 0) && (iy < 32);
      for (int dx = -1; dx <= 1; ++dx) {
        const int ix = ow + dx;
        const bool ok = rowok && (ix >= 0) && (ix < 32);
        const int tap = (dy + 1) * 3 + (dx + 1);
        const _Float16* arow = patch + ((dy + 1) * 18 + (mrow + dx + 1)) * 512;
        const _Float16* Bp = wpk + ((size_t)tap * 32 + nt) * 8192;
#pragma unroll 4
        for (int ks = 0; ks < 16; ++ks) {
          v16h a = fragA_f16_ptr(arow + ks * 32, ok);
          v16h b = fragB_pk(Bp + ks * 512);
          c = WMMA_F16(a, b, c);
        }
      }
    }
    const int n = nt * 16 + (lane & 15);
    const float inv = g[n] * rsqrtf(vv[n] + 1e-5f);
    const float sh = fb[n] * inv + bb[n] - mm[n] * inv;
#pragma unroll
    for (int r = 0; r < 8; ++r) {
      float val = c[r] * inv + sh;
      yout[(size_t)(mt * 16 + r + 8 * hi) * 512 + n] = val > 0.f ? val : 0.f;
    }
  }
}

// ---------------------------------------------------------------------------
// 5) Haar IDWT: dwt1 fp32 -> x_idwt f16 (B, 4096, 128)
// ---------------------------------------------------------------------------
__global__ void k_idwt(const float* __restrict__ xd, _Float16* __restrict__ out) {
  int idx = blockIdx.x * 256 + threadIdx.x;
  int cr = idx & 127; int p = idx >> 7;
  int j = p & 31; p >>= 5; int i = p & 31; int b = p >> 5;
  const float* s = xd + ((size_t)((b * 32 + i) * 32 + j)) * 512 + cr;
  float ll = s[0], lh = s[128], hl = s[256], hh = s[384];
  _Float16* o = out + ((size_t)(b * 4096 + (2 * i) * 64 + 2 * j)) * 128 + cr;
  o[0]              = (_Float16)(0.5f * (ll + lh + hl + hh));
  o[128]            = (_Float16)(0.5f * (ll + lh - hl - hh));
  o[64 * 128]       = (_Float16)(0.5f * (ll - lh + hl - hh));
  o[64 * 128 + 128] = (_Float16)(0.5f * (ll - lh - hl + hh));
}

// ---------------------------------------------------------------------------
// 6) 4x4 stride-4 conv -> kv tokens fp32 (1024, 512). M=1024 N=512 K=16*512
// ---------------------------------------------------------------------------
__global__ void k_kvqconv(const float* __restrict__ xd, const _Float16* __restrict__ wpk,
                          const float* __restrict__ wb, float* __restrict__ kvtok) {
  const int wave = threadIdx.x >> 5, lane = threadIdx.x & 31;
  const int mt = blockIdx.x, nt = blockIdx.y * 8 + wave;
  const int mrow = lane & 15, hi = lane >> 4;
  const int t = mt * 16 + mrow;
  const int b_ = t >> 6, oh = (t >> 3) & 7, ow = t & 7;
  v8f c = {};
  for (int dy = 0; dy < 4; ++dy)
    for (int dx = 0; dx < 4; ++dx) {
      const int tap = dy * 4 + dx;
      const float* arow = xd + ((size_t)((b_ * 32 + oh * 4 + dy) * 32 + ow * 4 + dx)) * 512;
      const _Float16* Bp = wpk + ((size_t)tap * 32 + nt) * 8192;
#pragma unroll 4
      for (int ks = 0; ks < 16; ++ks) {
        v16h a = fragA_f32_ptr(arow + ks * 32);
        v16h b = fragB_pk(Bp + ks * 512);
        c = WMMA_F16(a, b, c);
      }
    }
  const int n = nt * 16 + (lane & 15);
  const float bias = wb[n];
#pragma unroll
  for (int r = 0; r < 8; ++r)
    kvtok[(size_t)(mt * 16 + r + 8 * hi) * 512 + n] = c[r] + bias;
}

// ---------------------------------------------------------------------------
// 7) LayerNorm C=512, one wave per row; out f16
// ---------------------------------------------------------------------------
__global__ void k_ln(const float* __restrict__ in, const float* __restrict__ g,
                     const float* __restrict__ b, _Float16* __restrict__ out) {
  const int row = blockIdx.x * 8 + (threadIdx.x >> 5);
  const int lane = threadIdx.x & 31;
  const float* p = in + (size_t)row * 512;
  float s = 0.f, s2 = 0.f;
  for (int i = lane; i < 512; i += 32) { float v = p[i]; s += v; s2 += v * v; }
#pragma unroll
  for (int o = 16; o > 0; o >>= 1) { s += __shfl_xor(s, o, 32); s2 += __shfl_xor(s2, o, 32); }
  const float mu = s * (1.f / 512.f);
  const float inv = rsqrtf(s2 * (1.f / 512.f) - mu * mu + 1e-5f);
  _Float16* q = out + (size_t)row * 512;
  for (int i = lane; i < 512; i += 32) q[i] = (_Float16)((p[i] - mu) * inv * g[i] + b[i]);
}

// ---------------------------------------------------------------------------
// 8) kv GEMM (M=1024 N=1024 K=512); K/V written directly in packed-B fragment
//    layout per (b,head): S-GEMM needs B(d,lk), PV-GEMM needs B(lk,d).
// ---------------------------------------------------------------------------
__global__ void k_kvgemm(const _Float16* __restrict__ a_, const _Float16* __restrict__ wpk,
                         const float* __restrict__ kvb, _Float16* __restrict__ kpk,
                         _Float16* __restrict__ vpk) {
  const int wave = threadIdx.x >> 5, lane = threadIdx.x & 31;
  const int mt = blockIdx.x, nt = blockIdx.y * 8 + wave;
  const _Float16* A = a_ + (size_t)mt * 16 * 512;
  const _Float16* Bp = wpk + (size_t)nt * 8192;
  v8f c = {};
#pragma unroll 4
  for (int ks = 0; ks < 16; ++ks) {
    v16h a = fragA_f16(A, 512, ks * 32);
    v16h b = fragB_pk(Bp + ks * 512);
    c = WMMA_F16(a, b, c);
  }
  const int n = nt * 16 + (lane & 15), hi = lane >> 4;
  const float bias = kvb[n];
  const int c2 = n >> 9, head = (n >> 6) & 7, d = n & 63;
#pragma unroll
  for (int r = 0; r < 8; ++r) {
    int m = mt * 16 + r + 8 * hi;
    int b_ = m >> 6, lk = m & 63;
    size_t bh = (size_t)(b_ * 8 + head) * 4096;
    _Float16 val = (_Float16)(c[r] + bias);
    if (c2 == 0) {
      kpk[bh + (size_t)(lk >> 4) * 1024 + (size_t)(d >> 5) * 512 +
          (size_t)((lk & 15) + 16 * ((d >> 3) & 1)) * 16 + (d & 7) + 8 * ((d >> 4) & 1)] = val;
    } else {
      vpk[bh + (size_t)(d >> 4) * 1024 + (size_t)(lk >> 5) * 512 +
          (size_t)((d & 15) + 16 * ((lk >> 3) & 1)) * 16 + (lk & 7) + 8 * ((lk >> 4) & 1)] = val;
    }
  }
}

// ---------------------------------------------------------------------------
// 9) Attention: S = q·K^T (16x64/wave), softmax, ctx = P·V.
// ---------------------------------------------------------------------------
__global__ void k_attn(const _Float16* __restrict__ q, const _Float16* __restrict__ kpk,
                       const _Float16* __restrict__ vpk, float* __restrict__ attn_out,
                       _Float16* __restrict__ ctx_out) {
  __shared__ __align__(32) _Float16 plds[8][16 * 64];
  const int wave = threadIdx.x >> 5, lane = threadIdx.x & 31;
  const int n16 = lane & 15, hi = lane >> 4;
  const int bh = blockIdx.x >> 5;
  const int qtile = (blockIdx.x & 31) * 8 + wave;
  const _Float16* qbase = q + (size_t)bh * 4096 * 64 + (size_t)qtile * 16 * 64;
  const _Float16* kb = kpk + (size_t)bh * 4096;
  const _Float16* vb = vpk + (size_t)bh * 4096;

  v8f s[4];
#pragma unroll
  for (int f = 0; f < 4; ++f) {
    v8f c = {};
#pragma unroll
    for (int ks = 0; ks < 2; ++ks) {
      v16h a = fragA_f16(qbase, 64, ks * 32);
      v16h b = fragB_pk(kb + f * 1024 + ks * 512);
      c = WMMA_F16(a, b, c);
    }
    s[f] = c;
  }
  float rsum[8];
#pragma unroll
  for (int r = 0; r < 8; ++r) {
    float m = s[0][r];
#pragma unroll
    for (int f = 1; f < 4; ++f) m = fmaxf(m, s[f][r]);
#pragma unroll
    for (int o = 8; o > 0; o >>= 1) m = fmaxf(m, __shfl_xor(m, o, 32));
    float acc = 0.f;
#pragma unroll
    for (int f = 0; f < 4; ++f) { float e = __expf(s[f][r] - m); s[f][r] = e; acc += e; }
#pragma unroll
    for (int o = 8; o > 0; o >>= 1) acc += __shfl_xor(acc, o, 32);
    rsum[r] = 1.f / acc;
  }
  _Float16* pw = plds[wave];
#pragma unroll
  for (int f = 0; f < 4; ++f)
#pragma unroll
    for (int r = 0; r < 8; ++r) {
      float pv = s[f][r] * rsum[r];
      int row = r + 8 * hi, col = f * 16 + n16;
      pw[row * 64 + col] = (_Float16)pv;
      attn_out[((size_t)bh * 4096 + qtile * 16 + row) * 64 + col] = pv;
    }
  const int b_ = bh >> 3, head = bh & 7;
#pragma unroll
  for (int f = 0; f < 4; ++f) {
    v8f c = {};
#pragma unroll
    for (int ks = 0; ks < 2; ++ks) {
      v16h a = fragA_f16(pw, 64, ks * 32);
      v16h b = fragB_pk(vb + f * 1024 + ks * 512);
      c = WMMA_F16(a, b, c);
    }
#pragma unroll
    for (int r = 0; r < 8; ++r) {
      int m = qtile * 16 + r + 8 * hi;
      ctx_out[((size_t)b_ * 4096 + m) * 512 + head * 64 + f * 16 + n16] = (_Float16)c[r];
    }
  }
}

// ---------------------------------------------------------------------------
// 10) proj: out = concat(ctx f16 (M,512), idwt f16 (M,128)) @ proj_w.T + b
//     A tiles staged via async DMA; each wave serves 4 N-tiles.
// ---------------------------------------------------------------------------
__global__ void k_proj(const _Float16* __restrict__ ao, const _Float16* __restrict__ idw,
                       const _Float16* __restrict__ wpk, const float* __restrict__ pb,
                       float* __restrict__ out) {
  __shared__ __align__(16) _Float16 a1[16 * 512];
  __shared__ __align__(16) _Float16 a2[16 * 128];
  const int wave = threadIdx.x >> 5, lane = threadIdx.x & 31;
  const int mt = blockIdx.x;
  stage_async(ao + (size_t)mt * 8192, lds_off32(a1), 1024);
  stage_async(idw + (size_t)mt * 2048, lds_off32(a2), 256);
  wait_async_all();
  __syncthreads();
  const int hi = lane >> 4;
#pragma unroll
  for (int g = 0; g < 4; ++g) {
    const int nt = g * 8 + wave;
    const _Float16* Bp = wpk + (size_t)nt * 20 * 512;
    v8f c = {};
#pragma unroll 4
    for (int ks = 0; ks < 16; ++ks) {
      v16h a = fragA_f16(a1, 512, ks * 32);
      v16h b = fragB_pk(Bp + ks * 512);
      c = WMMA_F16(a, b, c);
    }
#pragma unroll
    for (int ks = 16; ks < 20; ++ks) {
      v16h a = fragA_f16(a2, 128, (ks - 16) * 32);
      v16h b = fragB_pk(Bp + ks * 512);
      c = WMMA_F16(a, b, c);
    }
    const int n = nt * 16 + (lane & 15);
    const float bias = pb[n];
#pragma unroll
    for (int r = 0; r < 8; ++r)
      out[(size_t)(mt * 16 + r + 8 * hi) * 512 + n] = c[r] + bias;
  }
}

// ---------------------------------------------------------------------------
extern "C" void kernel_launch(void* const* d_in, const int* in_sizes, int n_in,
                              void* d_out, int out_size, void* d_ws, size_t ws_size,
                              hipStream_t stream) {
  (void)in_sizes; (void)n_in; (void)out_size; (void)ws_size;
  const float* x      = (const float*)d_in[0];
  const float* red_w  = (const float*)d_in[1];
  const float* red_b  = (const float*)d_in[2];
  const float* bn1_g  = (const float*)d_in[3];
  const float* bn1_b  = (const float*)d_in[4];
  const float* bn1_m  = (const float*)d_in[5];
  const float* bn1_v  = (const float*)d_in[6];
  const float* filt_w = (const float*)d_in[7];
  const float* filt_b = (const float*)d_in[8];
  const float* bn2_g  = (const float*)d_in[9];
  const float* bn2_b  = (const float*)d_in[10];
  const float* bn2_m  = (const float*)d_in[11];
  const float* bn2_v  = (const float*)d_in[12];
  const float* kvq_w  = (const float*)d_in[13];
  const float* kvq_b  = (const float*)d_in[14];
  const float* q_w    = (const float*)d_in[15];
  const float* q_b    = (const float*)d_in[16];
  const float* ln_g   = (const float*)d_in[17];
  const float* ln_b   = (const float*)d_in[18];
  const float* kv_w   = (const float*)d_in[19];
  const float* kv_b   = (const float*)d_in[20];
  const float* proj_w = (const float*)d_in[21];
  const float* proj_b = (const float*)d_in[22];
  float* out = (float*)d_out;

  char* ws = (char*)d_ws;
  size_t off = 0;
  auto allocH = [&](size_t n) { _Float16* p = (_Float16*)(ws + off); off += ((n * 2 + 255) & ~(size_t)255); return p; };
  auto allocF = [&](size_t n) { float* p = (float*)(ws + off); off += ((n * 4 + 255) & ~(size_t)255); return p; };

  _Float16* x16   = allocH(33554432);     // x f16 (65536,512)
  _Float16* q_ws  = allocH(33554432);     // (B*8, 4096, 64) f16
  float*    xred  = allocF(8388608);      // (B,64,64,128) fp32
  _Float16* dwt0  = allocH(8388608);      // (B,32,32,512) f16 conv3 input
  float*    dwt1  = allocF(8388608);      // (B,32,32,512) fp32 post bn2/relu
  _Float16* idw   = allocH(8388608);      // (B,4096,128) f16
  float*    kvtok = allocF(524288);       // (1024,512) fp32
  _Float16* lnout = allocH(524288);       // (1024,512) f16
  _Float16* kpk   = allocH(524288);       // packed K frags
  _Float16* vpk   = allocH(524288);       // packed V frags
  _Float16* aout  = allocH(33554432);     // ctx (B,4096,512) f16
  _Float16* pk_q  = allocH(262144);
  _Float16* pk_r  = allocH(65536);
  _Float16* pk_f  = allocH(2359296);      // 9 taps
  _Float16* pk_kq = allocH(4194304);      // 16 taps
  _Float16* pk_kv = allocH(524288);
  _Float16* pk_p  = allocH(327680);       // KS=20

  dim3 blk(256);
  auto gsz = [](long n) { return dim3((unsigned)((n + 255) / 256)); };

  k_cvt4 <<<gsz(8388608), blk, 0, stream>>>((const float4*)x, (v4h*)x16, 8388608);
  k_packB<<<gsz(262144),  blk, 0, stream>>>(q_w,    pk_q, 16, 32, 1, 512, 1);
  k_packB<<<gsz(65536),   blk, 0, stream>>>(red_w,  pk_r, 16, 8, 1, 512, 1);
  k_packB<<<gsz(2359296), blk, 0, stream>>>(filt_w, pk_f, 16, 32, 9, 4608, 9);
  k_packB<<<gsz(4194304), blk, 0, stream>>>(kvq_w,  pk_kq, 16, 32, 16, 8192, 16);
  k_packB<<<gsz(524288),  blk, 0, stream>>>(kv_w,   pk_kv, 16, 64, 1, 512, 1);
  k_packB<<<gsz(327680),  blk, 0, stream>>>(proj_w, pk_p, 20, 32, 1, 640, 1);

  k_qproj  <<<dim3(4096),  blk, 0, stream>>>(x16, pk_q, q_b, q_ws);
  k_redconv<<<dim3(4096),  blk, 0, stream>>>(x16, pk_r, red_b, bn1_g, bn1_b, bn1_m, bn1_v, xred);
  k_dwt    <<<dim3(8192),  blk, 0, stream>>>(xred, dwt0);
  k_conv3  <<<dim3(1024),  blk, 0, stream>>>(dwt0, pk_f, filt_b, bn2_g, bn2_b, bn2_m, bn2_v, dwt1);
  k_idwt   <<<dim3(8192),  blk, 0, stream>>>(dwt1, idw);
  k_kvqconv<<<dim3(64, 4), blk, 0, stream>>>(dwt1, pk_kq, kvq_b, kvtok);
  k_ln     <<<dim3(128),   blk, 0, stream>>>(kvtok, ln_g, ln_b, lnout);
  k_kvgemm <<<dim3(64, 8), blk, 0, stream>>>(lnout, pk_kv, kv_b, kpk, vpk);
  k_attn   <<<dim3(4096),  blk, 0, stream>>>(q_ws, kpk, vpk, out + 33554432, aout);
  k_proj   <<<dim3(4096),  blk, 0, stream>>>(aout, idw, pk_p, proj_b, out);
}
